// NeRFOCCRenderer_53420803227895
// MI455X (gfx1250) — matrix operand order repaired and checked by reading, not compiled, not run on hardware
//
#include <hip/hip_runtime.h>
#include <hip/hip_bf16.h>
#include <math.h>

// ---------------------------------------------------------------------------
// NeRF occupancy renderer for MI455X (gfx1250, wave32, WMMA + TDM).
//   K1 mlp_kernel      : fused 3-layer density + rgb MLPs, all GEMM work on
//                        v_wmma_f32_16x16x32_f16 (incl. 64->1 / 64->3 layers
//                        as zero-padded N=16 tiles); weights staged via
//                        tensor_load_to_lds (TDM) and held in registers.
//   K2..K7             : zero accumulators, 3-kernel global scan, composite
//                        with wave-segmented reduction, background finalize.
// ---------------------------------------------------------------------------

#define S_TOTAL (1 << 20)   // 1048576 samples
#define R_TOTAL (1 << 15)   // 32768 rays
#define HID 64
#define HSTRIDE 72          // LDS row stride in halves (16B aligned, skewed)

typedef __attribute__((ext_vector_type(8)))  _Float16 v8h;
typedef __attribute__((ext_vector_type(16))) _Float16 v16h;
typedef __attribute__((ext_vector_type(8)))  float    v8f;
typedef __attribute__((ext_vector_type(4)))  unsigned v4u;
typedef __attribute__((ext_vector_type(8)))  unsigned v8u;

// ---- WMMA fragment loaders (layouts per CDNA5 ISA §7.12.2) -----

// A (16x32 f16): lane<16 -> M=lane, K in {k0..k0+7, k0+16..k0+23};
// lane>=16 -> M=lane-16, K in {k0+8..k0+15, k0+24..k0+31}.
__device__ inline v16h ld_frag_A(const _Float16* h, int k0) {
  int lane = threadIdx.x & 31;
  const _Float16* p = h + (lane & 15) * HSTRIDE + k0 + ((lane >> 4) << 3);
  v8h lo = *(const v8h*)p;
  v8h hi = *(const v8h*)(p + 16);
  v16h r;
#pragma unroll
  for (int j = 0; j < 8; ++j) { r[j] = lo[j]; r[j + 8] = hi[j]; }
  return r;
}

// B (32x16 f16): lane<16 -> N=n0+lane, K in {k0..k0+7, k0+16..k0+23}; lane>=16 ->
// N=n0+lane-16, K in {k0+8..k0+15, k0+24..k0+31}. Weights staged TRANSPOSED in LDS
// as wt[N][K] so each lane's 16 K-values are two b128 DS loads.
__device__ inline v16h ld_frag_B(const _Float16* wt, int k0, int n0) {
  int lane = threadIdx.x & 31;
  const _Float16* p = wt + (n0 + (lane & 15)) * HSTRIDE + k0 + ((lane >> 4) << 3);
  v8h lo = *(const v8h*)p;
  v8h hi = *(const v8h*)(p + 16);
  v16h r;
#pragma unroll
  for (int j = 0; j < 8; ++j) { r[j] = lo[j]; r[j + 8] = hi[j]; }
  return r;
}

__device__ inline v8f wmma_f16(v16h a, v16h b, v8f c) {
  return __builtin_amdgcn_wmma_f32_16x16x32_f16(false, a, false, b,
                                                (short)0, c, false, false);
}

// Issue a TDM load of one contiguous row of `n_elts` f32 from `gptr` into LDS
// at byte offset `lds_addr` (D# per CDNA5 ISA).
__device__ inline void tdm_load_f32_row(const float* gptr, unsigned lds_addr,
                                        unsigned n_elts) {
  unsigned long long ga = (unsigned long long)(__SIZE_TYPE__)gptr;
  v4u g0;
  g0[0] = 1u;                                   // count=1 (valid descriptor)
  g0[1] = lds_addr;                             // lds_addr (bytes)
  g0[2] = (unsigned)ga;                         // global_addr[31:0]
  g0[3] = (unsigned)((ga >> 32) & 0x1FFFFFFu) | (2u << 30);  // addr[56:32]|type=2
  v8u g1;
  g1[0] = (2u << 16);                           // wg_mask=0, data_size=2 (4B)
  g1[1] = (n_elts & 0xFFFFu) << 16;             // tensor_dim0[15:0]
  g1[2] = ((n_elts >> 16) & 0xFFFFu) | (1u << 16);  // dim0[31:16] | dim1=1
  g1[3] = (n_elts & 0xFFFFu) << 16;             // tile_dim0 = n_elts
  g1[4] = 1u;                                   // tile_dim1 = 1
  g1[5] = n_elts;                               // dim0_stride[31:0]
  g1[6] = (n_elts & 0xFFFFu) << 16;             // dim1_stride[15:0] | stride0 hi
  g1[7] = 0u;
  asm volatile("tensor_load_to_lds %0, %1" :: "s"(g0), "s"(g1) : "memory");
}

// ---------------------------------------------------------------------------
// K1: fused MLP kernel. 128 threads = 4 independent waves; each wave owns a
// 16-sample tile per iteration. 20 WMMAs per tile (8+2 per MLP).
// ---------------------------------------------------------------------------
struct SMemMLP {
  _Float16 sWd2[64 * HSTRIDE];   // transposed [N][K] f16
  _Float16 sWr2[64 * HSTRIDE];
  _Float16 sW3d[16 * HSTRIDE];   // layer-3 density, zero-padded to N=16
  _Float16 sW3r[16 * HSTRIDE];   // layer-3 rgb, cols 0..2 live
  float sWd1[192], sBd1[64];
  float sWr1[192], sBr1[64];
  float sBd2[64], sBr2[64], sBd3, sBr3[3];
  union {
    float stage[4096];           // 16KB TDM staging (one 64x64 f32 matrix)
    struct {
      _Float16 hA[4][16 * HSTRIDE];
      _Float16 hB[4][16 * HSTRIDE];
      float cbuf[4][16 * 4];     // per-wave layer-3 column scratch
    } run;
  } u;
};

__global__ __launch_bounds__(128, 1) void mlp_kernel(
    const float* __restrict__ rays_o, const float* __restrict__ rays_d,
    const float* __restrict__ t_starts, const float* __restrict__ t_ends,
    const int* __restrict__ ray_indices,
    const float* __restrict__ W_d1, const float* __restrict__ b_d1,
    const float* __restrict__ W_d2, const float* __restrict__ b_d2,
    const float* __restrict__ W_d3, const float* __restrict__ b_d3,
    const float* __restrict__ W_r1, const float* __restrict__ b_r1,
    const float* __restrict__ W_r2, const float* __restrict__ b_r2,
    const float* __restrict__ W_r3, const float* __restrict__ b_r3,
    float* __restrict__ sd_out, float* __restrict__ rgb_out) {
  __shared__ SMemMLP sm;

  const int t = threadIdx.x;
  const int wv = t >> 5, lane = t & 31;
  const unsigned stage_lds =
      (unsigned)__builtin_amdgcn_readfirstlane((int)(__SIZE_TYPE__)(void*)sm.u.stage);

  // ---- small weights / biases (plain loads) ----
  for (int i = t; i < 192; i += 128) { sm.sWd1[i] = W_d1[i]; sm.sWr1[i] = W_r1[i]; }
  for (int i = t; i < 64; i += 128) {
    sm.sBd1[i] = b_d1[i]; sm.sBr1[i] = b_r1[i];
    sm.sBd2[i] = b_d2[i]; sm.sBr2[i] = b_r2[i];
  }
  // layer-3 tiles, transposed + zero padded to N=16
  for (int i = t; i < 16 * 64; i += 128) {
    int n = i >> 6, k = i & 63;
    sm.sW3d[n * HSTRIDE + k] = (n == 0) ? (_Float16)W_d3[k] : (_Float16)0.f;
    sm.sW3r[n * HSTRIDE + k] = (n < 3) ? (_Float16)W_r3[k * 3 + n] : (_Float16)0.f;
  }
  if (t == 0) sm.sBd3 = b_d3[0];
  if (t < 3) sm.sBr3[t] = b_r3[t];

  // ---- TDM-stage the two 64x64 f32 matrices, convert to transposed f16 ----
  if (t < 32) {                       // wave 0 issues the DMA (EXEC ignored)
    tdm_load_f32_row(W_d2, stage_lds, 4096);
    __builtin_amdgcn_s_wait_tensorcnt(0);
  }
  __syncthreads();
  for (int i = t; i < 4096; i += 128) {
    int k = i >> 6, n = i & 63;       // source row-major [K][N]
    sm.sWd2[n * HSTRIDE + k] = (_Float16)sm.u.stage[i];
  }
  __syncthreads();
  if (t < 32) {
    tdm_load_f32_row(W_r2, stage_lds, 4096);
    __builtin_amdgcn_s_wait_tensorcnt(0);
  }
  __syncthreads();
  for (int i = t; i < 4096; i += 128) {
    int k = i >> 6, n = i & 63;
    sm.sWr2[n * HSTRIDE + k] = (_Float16)sm.u.stage[i];
  }
  __syncthreads();

  // ---- hoist all B fragments into registers (loop-invariant weights) ----
  v16h bd[2][4], br[2][4], b3d[2], b3r[2];
#pragma unroll
  for (int kt = 0; kt < 2; ++kt) {
#pragma unroll
    for (int nt = 0; nt < 4; ++nt) {
      bd[kt][nt] = ld_frag_B(sm.sWd2, kt * 32, nt * 16);
      br[kt][nt] = ld_frag_B(sm.sWr2, kt * 32, nt * 16);
    }
    b3d[kt] = ld_frag_B(sm.sW3d, kt * 32, 0);
    b3r[kt] = ld_frag_B(sm.sW3r, kt * 32, 0);
  }

  _Float16* hAw = sm.u.run.hA[wv];
  _Float16* hBw = sm.u.run.hB[wv];
  float* cbw = sm.u.run.cbuf[wv];
  const int nWaves = gridDim.x * 4;
  const int nTiles = S_TOTAL / 16;
  const int nCol = lane & 15;
  const int mrow = (lane >> 4) << 3;

  for (int tile = blockIdx.x * 4 + wv; tile < nTiles; tile += nWaves) {
    // ---- per-sample setup (lanes 0..15 own samples; 16..31 mirror) ----
    int s = tile * 16 + (lane & 15);
    int ri = ray_indices[s];
    float ts = t_starts[s], te = t_ends[s];
    float mid = 0.5f * (ts + te), dt = te - ts;
    float px = rays_o[ri * 3 + 0] + rays_d[ri * 3 + 0] * mid;
    float py = rays_o[ri * 3 + 1] + rays_d[ri * 3 + 1] * mid;
    float pz = rays_o[ri * 3 + 2] + rays_d[ri * 3 + 2] * mid;
    {   // prefetch next tile's sample stream (global_prefetch_b8)
      int sn = s + nWaves * 16;
      if (sn < S_TOTAL) {
        __builtin_prefetch(t_starts + sn, 0, 3);
        __builtin_prefetch(t_ends + sn, 0, 3);
        __builtin_prefetch(ray_indices + sn, 0, 3);
      }
    }

    // broadcast position of sample (lane>>1) for layer-1 work split
    int smp = lane >> 1;
    float qx = __shfl(px, smp, 32), qy = __shfl(py, smp, 32), qz = __shfl(pz, smp, 32);
    int fb = (lane & 1) * 32;

    // =============== density MLP ===============
#pragma unroll 4
    for (int j = 0; j < 32; ++j) {   // layer 1: 3->64 (VALU), relu -> hA f16
      int f = fb + j;
      float v = qx * sm.sWd1[f] + qy * sm.sWd1[64 + f] + qz * sm.sWd1[128 + f] + sm.sBd1[f];
      hAw[smp * HSTRIDE + f] = (_Float16)(v > 0.f ? v : 0.f);
    }
    {   // layer 2: 64->64 WMMA, bias+relu -> hB f16
      v16h a0 = ld_frag_A(hAw, 0);
      v16h a1 = ld_frag_A(hAw, 32);
#pragma unroll
      for (int nt = 0; nt < 4; ++nt) {
        v8f acc = {0.f, 0.f, 0.f, 0.f, 0.f, 0.f, 0.f, 0.f};
        acc = wmma_f16(a0, bd[0][nt], acc);
        acc = wmma_f16(a1, bd[1][nt], acc);
        float bias = sm.sBd2[nt * 16 + nCol];
#pragma unroll
        for (int r8 = 0; r8 < 8; ++r8) {
          float v = acc[r8] + bias;
          hBw[(mrow + r8) * HSTRIDE + nt * 16 + nCol] = (_Float16)(v > 0.f ? v : 0.f);
        }
      }
    }
    {   // layer 3: 64->1 as zero-padded WMMA tile; softplus; sigma*dt
      v16h a0 = ld_frag_A(hBw, 0);
      v16h a1 = ld_frag_A(hBw, 32);
      v8f acc = {0.f, 0.f, 0.f, 0.f, 0.f, 0.f, 0.f, 0.f};
      acc = wmma_f16(a0, b3d[0], acc);
      acc = wmma_f16(a1, b3d[1], acc);
      if (nCol == 0) {
#pragma unroll
        for (int r8 = 0; r8 < 8; ++r8) cbw[(mrow + r8) * 4] = acc[r8];
      }
      if (lane < 16) {   // wave-internal DS ordering: stores above are visible
        float x = cbw[lane * 4] + sm.sBd3;
        float sp = (x > 15.f) ? x : log1pf(__expf(x));
        sd_out[s] = sp * dt;
      }
    }

    // =============== rgb MLP ===============
#pragma unroll 4
    for (int j = 0; j < 32; ++j) {
      int f = fb + j;
      float v = qx * sm.sWr1[f] + qy * sm.sWr1[64 + f] + qz * sm.sWr1[128 + f] + sm.sBr1[f];
      hAw[smp * HSTRIDE + f] = (_Float16)(v > 0.f ? v : 0.f);
    }
    {
      v16h a0 = ld_frag_A(hAw, 0);
      v16h a1 = ld_frag_A(hAw, 32);
#pragma unroll
      for (int nt = 0; nt < 4; ++nt) {
        v8f acc = {0.f, 0.f, 0.f, 0.f, 0.f, 0.f, 0.f, 0.f};
        acc = wmma_f16(a0, br[0][nt], acc);
        acc = wmma_f16(a1, br[1][nt], acc);
        float bias = sm.sBr2[nt * 16 + nCol];
#pragma unroll
        for (int r8 = 0; r8 < 8; ++r8) {
          float v = acc[r8] + bias;
          hBw[(mrow + r8) * HSTRIDE + nt * 16 + nCol] = (_Float16)(v > 0.f ? v : 0.f);
        }
      }
    }
    {   // layer 3: 64->3 as zero-padded WMMA tile; sigmoid; planar store
      v16h a0 = ld_frag_A(hBw, 0);
      v16h a1 = ld_frag_A(hBw, 32);
      v8f acc = {0.f, 0.f, 0.f, 0.f, 0.f, 0.f, 0.f, 0.f};
      acc = wmma_f16(a0, b3r[0], acc);
      acc = wmma_f16(a1, b3r[1], acc);
      if (nCol < 3) {
#pragma unroll
        for (int r8 = 0; r8 < 8; ++r8) cbw[(mrow + r8) * 4 + nCol] = acc[r8];
      }
      if (lane < 16) {
#pragma unroll
        for (int c = 0; c < 3; ++c) {
          float x = cbw[lane * 4 + c] + sm.sBr3[c];
          rgb_out[c * S_TOTAL + s] = 1.f / (1.f + __expf(-x));
        }
      }
    }
  }
}

// ---------------------------------------------------------------------------
// K2: zero accumulators
__global__ void zero_acc(float* __restrict__ p, int n) {
  int i = blockIdx.x * 256 + threadIdx.x;
  if (i < n) p[i] = 0.f;
}

// K3: chunked inclusive scan; 4096 elems/block, 16/thread, float4 I/O
__global__ __launch_bounds__(256) void scan_chunk(const float* __restrict__ sd,
                                                  float* __restrict__ lscan,
                                                  float* __restrict__ csum) {
  __shared__ float tsum[256];
  int t = threadIdx.x;
  int base = blockIdx.x * 4096 + t * 16;
  float v[16];
  const float4* p4 = (const float4*)(sd + base);
#pragma unroll
  for (int i = 0; i < 4; ++i) {
    float4 q = p4[i];
    v[4 * i] = q.x; v[4 * i + 1] = q.y; v[4 * i + 2] = q.z; v[4 * i + 3] = q.w;
  }
  float run = 0.f;
#pragma unroll
  for (int i = 0; i < 16; ++i) { run += v[i]; v[i] = run; }
  tsum[t] = run;
  __syncthreads();
  float x = run;
  for (int off = 1; off < 256; off <<= 1) {
    float y = (t >= off) ? tsum[t - off] : 0.f;
    __syncthreads();
    x += y;
    tsum[t] = x;
    __syncthreads();
  }
  float prefix = x - run;
  float4* o4 = (float4*)(lscan + base);
#pragma unroll
  for (int i = 0; i < 4; ++i) {
    float4 q;
    q.x = v[4 * i] + prefix; q.y = v[4 * i + 1] + prefix;
    q.z = v[4 * i + 2] + prefix; q.w = v[4 * i + 3] + prefix;
    o4[i] = q;
  }
  if (t == 255) csum[blockIdx.x] = x;
}

// K4: exclusive scan of 256 chunk totals (single block)
__global__ __launch_bounds__(256) void scan_tops(const float* __restrict__ csum,
                                                 float* __restrict__ coff) {
  __shared__ float sh[256];
  int t = threadIdx.x;
  float own = csum[t];
  sh[t] = own;
  __syncthreads();
  float x = own;
  for (int off = 1; off < 256; off <<= 1) {
    float y = (t >= off) ? sh[t - off] : 0.f;
    __syncthreads();
    x += y;
    sh[t] = x;
    __syncthreads();
  }
  coff[t] = x - own;
}

// K5: global exclusive scan in-place; record per-ray start offset
__global__ void finalize_scan(const float* __restrict__ sd, float* __restrict__ lscan,
                              const float* __restrict__ coff, const int* __restrict__ ridx,
                              float* __restrict__ rstart) {
  int s = blockIdx.x * 256 + threadIdx.x;
  float excl = lscan[s] - sd[s] + coff[s >> 12];
  lscan[s] = excl;
  int r = ridx[s];
  if (s == 0 || ridx[s - 1] != r) rstart[r] = excl;
}

// K6: weights + wave-segmented reduction (ray_indices sorted) + atomics
__global__ __launch_bounds__(256) void composite(
    const float* __restrict__ sd, const float* __restrict__ excl,
    const float* __restrict__ rstart, const int* __restrict__ ridx,
    const float* __restrict__ ts, const float* __restrict__ te,
    const float* __restrict__ rgb,
    float* __restrict__ accO, float* __restrict__ accD, float* __restrict__ accC) {
  int s = blockIdx.x * 256 + threadIdx.x;
  int lane = threadIdx.x & 31;
  int r = ridx[s];
  float sdt = sd[s];
  float trans = __expf(-(excl[s] - rstart[r]));
  float alpha = 1.f - __expf(-sdt);
  float w = trans * alpha;
  float mid = 0.5f * (ts[s] + te[s]);
  float v0 = w, v1 = w * mid;
  float v2 = w * rgb[s], v3 = w * rgb[S_TOTAL + s], v4 = w * rgb[2 * S_TOTAL + s];
#pragma unroll
  for (int off = 1; off < 32; off <<= 1) {
    int ru = __shfl_up(r, off, 32);
    float u0 = __shfl_up(v0, off, 32), u1 = __shfl_up(v1, off, 32);
    float u2 = __shfl_up(v2, off, 32), u3 = __shfl_up(v3, off, 32);
    float u4 = __shfl_up(v4, off, 32);
    if (lane >= off && ru == r) { v0 += u0; v1 += u1; v2 += u2; v3 += u3; v4 += u4; }
  }
  int rd = __shfl_down(r, 1, 32);
  if (lane == 31 || rd != r) {  // segment tail: one atomic set per ray-run
    atomicAdd(accO + r, v0);
    atomicAdd(accD + r, v1);
    atomicAdd(accC + r, v2);
    atomicAdd(accC + R_TOTAL + r, v3);
    atomicAdd(accC + 2 * R_TOTAL + r, v4);
  }
}

// K7: background composition + output layout (colors[R,3], dist[R], opac[R])
__global__ void finalize_rays(const float* __restrict__ accO, const float* __restrict__ accD,
                              const float* __restrict__ accC, float* __restrict__ out) {
  int r = blockIdx.x * 256 + threadIdx.x;
  float o = accO[r];
  float rest = 1.f - o;
  out[r * 3 + 0] = accC[r] + 0.5f * rest;
  out[r * 3 + 1] = accC[R_TOTAL + r] + 0.5f * rest;
  out[r * 3 + 2] = accC[2 * R_TOTAL + r] + 0.5f * rest;
  out[3 * R_TOTAL + r] = accD[r] + 5.f * rest;
  out[4 * R_TOTAL + r] = o;
}

// ---------------------------------------------------------------------------
extern "C" void kernel_launch(void* const* d_in, const int* in_sizes, int n_in,
                              void* d_out, int out_size, void* d_ws, size_t ws_size,
                              hipStream_t stream) {
  const float* rays_o   = (const float*)d_in[0];
  const float* rays_d   = (const float*)d_in[1];
  const float* t_starts = (const float*)d_in[2];
  const float* t_ends   = (const float*)d_in[3];
  const int*   ridx     = (const int*)d_in[4];
  const float* W_d1 = (const float*)d_in[5];  const float* b_d1 = (const float*)d_in[6];
  const float* W_d2 = (const float*)d_in[7];  const float* b_d2 = (const float*)d_in[8];
  const float* W_d3 = (const float*)d_in[9];  const float* b_d3 = (const float*)d_in[10];
  const float* W_r1 = (const float*)d_in[11]; const float* b_r1 = (const float*)d_in[12];
  const float* W_r2 = (const float*)d_in[13]; const float* b_r2 = (const float*)d_in[14];
  const float* W_r3 = (const float*)d_in[15]; const float* b_r3 = (const float*)d_in[16];
  float* out = (float*)d_out;

  // workspace (floats): sd[S] | rgb[3S planar] | lscan[S] | csum[256] |
  // coff[256] | rstart[R] | accO[R] | accD[R] | accC[3R]   (~21.8 MB)
  float* ws     = (float*)d_ws;
  float* sd     = ws;
  float* rgb    = ws + S_TOTAL;
  float* lscan  = ws + 4 * (size_t)S_TOTAL;
  float* csum   = ws + 5 * (size_t)S_TOTAL;
  float* coff   = csum + 256;
  float* rstart = coff + 256;
  float* accO   = rstart + R_TOTAL;
  float* accD   = accO + R_TOTAL;
  float* accC   = accD + R_TOTAL;

  mlp_kernel<<<512, 128, 0, stream>>>(rays_o, rays_d, t_starts, t_ends, ridx,
                                      W_d1, b_d1, W_d2, b_d2, W_d3, b_d3,
                                      W_r1, b_r1, W_r2, b_r2, W_r3, b_r3,
                                      sd, rgb);
  zero_acc<<<(5 * R_TOTAL + 255) / 256, 256, 0, stream>>>(accO, 5 * R_TOTAL);
  scan_chunk<<<S_TOTAL / 4096, 256, 0, stream>>>(sd, lscan, csum);
  scan_tops<<<1, 256, 0, stream>>>(csum, coff);
  finalize_scan<<<S_TOTAL / 256, 256, 0, stream>>>(sd, lscan, coff, ridx, rstart);
  composite<<<S_TOTAL / 256, 256, 0, stream>>>(sd, lscan, rstart, ridx,
                                               t_starts, t_ends, rgb,
                                               accO, accD, accC);
  finalize_rays<<<R_TOTAL / 256, 256, 0, stream>>>(accO, accD, accC, out);
}
